// RGCNEncoder_9105330668028
// MI455X (gfx1250) — compile-verified, matrix operand order checked by reading, not run on hardware
//
#include <hip/hip_runtime.h>
#include <hip/hip_bf16.h>
#include <stdint.h>

// ---------------------------------------------------------------------------
// Problem constants (fixed by the reference)
// ---------------------------------------------------------------------------
constexpr int NN    = 4096;   // nodes
constexpr int FIN   = 4096;   // input features
constexpr int HD    = 128;    // hidden = emb = 128
constexpr int RR    = 8;      // relations
constexpr int GG    = 3;      // groups
constexpr int EE    = 65536;  // edges
constexpr int NS    = (RR + 1) * HD;  // 1152: stacked GEMM width (8 relations + root)
constexpr float LAMDA = 0.01f;
constexpr float EPSV  = 1e-5f;

// ---------------------------------------------------------------------------
// Vector types for WMMA
// ---------------------------------------------------------------------------
typedef __attribute__((ext_vector_type(16))) __bf16   v16bf_t;
typedef __attribute__((ext_vector_type(8)))  float    v8f_t;
typedef __attribute__((ext_vector_type(4)))  uint32_t v4u_t;
typedef __attribute__((ext_vector_type(2)))  uint32_t v2u_t;
typedef __attribute__((ext_vector_type(4)))  float    v4f_t;

union Frag16 { v4u_t u[2]; v16bf_t v; };

// s_wait_asynccnt: builtin if present, else raw SOPP
#if defined(__has_builtin)
#  if __has_builtin(__builtin_amdgcn_s_wait_asynccnt)
#    define WAIT_ASYNC(n) __builtin_amdgcn_s_wait_asynccnt(n)
#  endif
#endif
#ifndef WAIT_ASYNC
#  define WAIT_ASYNC(n) asm volatile("s_wait_asynccnt %0" :: "i"(n) : "memory")
#endif

// CDNA5 async copy: global memory -> LDS, 16B per lane, tracked by ASYNCcnt
__device__ __forceinline__ void async_ld_b128(uint32_t lds_off, uint64_t gaddr) {
  asm volatile("global_load_async_to_lds_b128 %0, %1, off"
               :: "v"(lds_off), "v"(gaddr) : "memory");
}

__device__ __forceinline__ uint16_t f2bf(float f) {
  union { float f; uint32_t u; } c; c.f = f;
  uint32_t u = c.u;
  u += 0x7fffu + ((u >> 16) & 1u);  // round-to-nearest-even
  return (uint16_t)(u >> 16);
}

// ---------------------------------------------------------------------------
// Utility kernels
// ---------------------------------------------------------------------------
__global__ void zero_f32_kernel(float* __restrict__ p, int count) {
  int i = blockIdx.x * blockDim.x + threadIdx.x;
  if (i < count) p[i] = 0.0f;
}

// fp32 -> bf16, 4 elements per thread (count must be multiple of 4)
__global__ void cast_bf16_kernel(const float* __restrict__ src,
                                 uint16_t* __restrict__ dst, int count4) {
  int i = blockIdx.x * blockDim.x + threadIdx.x;
  if (i >= count4) return;
  v4f_t v = *(const v4f_t*)(src + (size_t)i * 4);
  v2u_t o;
  o.x = (uint32_t)f2bf(v.x) | ((uint32_t)f2bf(v.y) << 16);
  o.y = (uint32_t)f2bf(v.z) | ((uint32_t)f2bf(v.w) << 16);
  *(v2u_t*)(dst + (size_t)i * 4) = o;
}

// Build TRANSPOSED stacked B matrix BT[NS x K] bf16 (K contiguous):
// rows [0,1024) = W[r][k][h] with row = r*128+h, rows [1024,1152) = root[k][h]
__global__ void pack_stackT_kernel(const float* __restrict__ W,
                                   const float* __restrict__ root,
                                   uint16_t* __restrict__ out, int K) {
  int idx = blockIdx.x * blockDim.x + threadIdx.x;  // col fastest -> coalesced W reads
  if (idx >= NS * K) return;
  int k   = idx / NS;
  int col = idx - k * NS;
  float v;
  if (col < RR * HD) {
    int r = col >> 7;
    int h = col & 127;
    v = W[((size_t)r * K + k) * HD + h];
  } else {
    v = root[(size_t)k * HD + (col - RR * HD)];
  }
  out[(size_t)col * K + k] = f2bf(v);
}

// Count edges per (dst, rel) segment
__global__ void count_edges_kernel(const int* __restrict__ edge_index,
                                   const int* __restrict__ edge_type,
                                   float* __restrict__ cnt) {
  int e = blockIdx.x * blockDim.x + threadIdx.x;
  if (e >= EE) return;
  int dst = edge_index[EE + e];
  int r   = edge_type[e];
  atomicAdd(&cnt[dst * RR + r], 1.0f);
}

// Scatter-mean in the embedded (128-wide) space: acc[dst] += Y[src, r*128 + h] / cnt
__global__ void scatter_kernel(const float* __restrict__ Y,
                               const int* __restrict__ edge_index,
                               const int* __restrict__ edge_type,
                               const float* __restrict__ cnt,
                               float* __restrict__ acc) {
  int t = blockIdx.x * blockDim.x + threadIdx.x;   // EE * 32 threads
  if (t >= EE * 32) return;
  int e  = t >> 5;
  int h0 = (t & 31) << 2;
  int src = edge_index[e];
  int dst = edge_index[EE + e];
  int r   = edge_type[e];
  float w = 1.0f / fmaxf(cnt[dst * RR + r], 1.0f);
  v4f_t y = *(const v4f_t*)(Y + (size_t)src * NS + r * HD + h0);
  float* a = acc + (size_t)dst * HD + h0;
  atomicAdd(a + 0, y.x * w);
  atomicAdd(a + 1, y.y * w);
  atomicAdd(a + 2, y.z * w);
  atomicAdd(a + 3, y.w * w);
}

// out[n,h] = acc[n,h] + Y[n, 1024+h] (root term) + bias[h]
__global__ void finish_kernel(const float* __restrict__ acc,
                              const float* __restrict__ Y,
                              const float* __restrict__ bias,
                              float* __restrict__ out) {
  int idx = blockIdx.x * blockDim.x + threadIdx.x;   // NN * HD
  if (idx >= NN * HD) return;
  int n = idx >> 7;
  int h = idx & 127;
  out[idx] = acc[idx] + Y[(size_t)n * NS + RR * HD + h] + bias[h];
}

// ---------------------------------------------------------------------------
// bf16 WMMA GEMM: C[NN x NS] = A[NN x K] * BT[NS x K]^T, fp32 accumulate.
// Block tile 128x128, BK=32, 128 threads (4 waves, each a 64x64 tile:
// 16 WMMA per 16 ds_load_b128 -> 1:1 compute/LDS-read ratio).
// Both A and BT are K-major, so staging is pure contiguous 16B async copies.
// Double-buffered LDS + GLOBAL_LOAD_ASYNC_TO_LDS_B128 software pipeline
// with the final iteration peeled for branch-free steady state.
// Requires K%32==0, K>=64.
// ---------------------------------------------------------------------------
#define BM 128
#define BN 128
#define BKK 32
#define AS_LD 40   // padded K-stride (elements); 40*2=80B keeps 16B alignment

__global__ __launch_bounds__(128)
void gemm_bf16_wmma_kernel(const uint16_t* __restrict__ A,
                           const uint16_t* __restrict__ BT,
                           float* __restrict__ C, int K) {
  __shared__ uint16_t As[2][BM * AS_LD];  // [buf][row][k]
  __shared__ uint16_t Bs[2][BN * AS_LD];  // [buf][col][k]

  const int tid  = threadIdx.x;
  const int lane = tid & 31;
  const int wave = tid >> 5;        // 0..3
  const int wm = (wave & 1) * 64;   // wave's M offset inside block tile
  const int wn = (wave >> 1) * 64;  // wave's N offset inside block tile
  const int mBlk = blockIdx.y * BM;
  const int nBlk = blockIdx.x * BN;

  const int r16 = lane & 15;
  const int kg  = (lane >> 4) << 3;   // 0 or 8: K-half select per ISA A-layout

  // staging: 512 16B-chunks per matrix tile, 4 per thread per matrix
  const int rbase = tid >> 2;         // 0..31
  const int c8    = (tid & 3) << 3;   // k offset within tile: 0/8/16/24

  v8f_t acc[4][4];
  const v8f_t vz = {0.f,0.f,0.f,0.f,0.f,0.f,0.f,0.f};
#pragma unroll
  for (int i = 0; i < 4; ++i)
#pragma unroll
    for (int j = 0; j < 4; ++j) acc[i][j] = vz;

  // issue one tile's async copies: 8 per thread (4 A + 4 B)
  auto issue_tile = [&](int k0, int buf) {
#pragma unroll
    for (int q = 0; q < 4; ++q) {
      int row = q * 32 + rbase;
      async_ld_b128((uint32_t)(uintptr_t)&As[buf][row * AS_LD + c8],
                    (uint64_t)(uintptr_t)(A + (size_t)(mBlk + row) * K + k0 + c8));
    }
#pragma unroll
    for (int q = 0; q < 4; ++q) {
      int row = q * 32 + rbase;
      async_ld_b128((uint32_t)(uintptr_t)&Bs[buf][row * AS_LD + c8],
                    (uint64_t)(uintptr_t)(BT + (size_t)(nBlk + row) * K + k0 + c8));
    }
  };

  // 16 WMMA from buffer `buf`; fragment loads per ISA 16-bit A layout:
  // lane holds row r16, K-half kg; VGPRs 0-3 = K kg..kg+7,
  // VGPRs 4-7 = K 16+kg..16+kg+7 -> two ds_load_b128 per fragment.
  auto compute_tile = [&](int buf) {
    Frag16 afr[4], bfr[4];
#pragma unroll
    for (int i = 0; i < 4; ++i) {
      const uint16_t* p = &As[buf][(wm + i * 16 + r16) * AS_LD + kg];
      afr[i].u[0] = *(const v4u_t*)p;
      afr[i].u[1] = *(const v4u_t*)(p + 16);
    }
#pragma unroll
    for (int j = 0; j < 4; ++j) {
      const uint16_t* p = &Bs[buf][(wn + j * 16 + r16) * AS_LD + kg];
      bfr[j].u[0] = *(const v4u_t*)p;
      bfr[j].u[1] = *(const v4u_t*)(p + 16);
    }
#pragma unroll
    for (int i = 0; i < 4; ++i)
#pragma unroll
      for (int j = 0; j < 4; ++j)
        acc[i][j] = __builtin_amdgcn_wmma_f32_16x16x32_bf16(
            false, afr[i].v, false, bfr[j].v, (short)0, acc[i][j], false, false);
  };

  const int nIter = K / BKK;
  issue_tile(0, 0);
  int cur = 0;

  // steady state: branch-free issue -> wait older 8 -> barrier -> 16 WMMA
  for (int it = 0; it < nIter - 1; ++it) {
    issue_tile((it + 1) * BKK, cur ^ 1);  // buffer freed by trailing barrier of prev iter
    WAIT_ASYNC(8);                        // async loads complete in order: retire tile `it`
    __syncthreads();
    compute_tile(cur);
    __syncthreads();
    cur ^= 1;
  }
  // peeled final iteration
  WAIT_ASYNC(0);
  __syncthreads();
  compute_tile(cur);

  // Store C: 16x16 f32 D layout — lane%16 = N col, VGPR v = row v + 8*(lane/16).
  // NS is constexpr so v*NS folds into immediate store offsets.
  const int rowHalf = (lane >> 4) * 8;
#pragma unroll
  for (int i = 0; i < 4; ++i) {
#pragma unroll
    for (int j = 0; j < 4; ++j) {
      float* cp = C + (size_t)(mBlk + wm + i * 16 + rowHalf) * NS
                    + (nBlk + wn + j * 16 + r16);
#pragma unroll
      for (int v = 0; v < 8; ++v)
        cp[(size_t)v * NS] = acc[i][j][v];
    }
  }
}

// ---------------------------------------------------------------------------
// DiffGroupNorm pieces
// ---------------------------------------------------------------------------
__global__ void dgn_softmax_kernel(const float* __restrict__ h,
                                   const float* __restrict__ lin_w,
                                   const float* __restrict__ lin_b,
                                   float* __restrict__ s) {
  int wave = threadIdx.x >> 5;
  int lane = threadIdx.x & 31;
  int node = blockIdx.x * 8 + wave;
  if (node >= NN) return;
  const float* row = h + (size_t)node * HD;
  float d0 = 0.f, d1 = 0.f, d2 = 0.f;
  for (int t = lane; t < HD; t += 32) {
    float xv = row[t];
    d0 += xv * lin_w[t * GG + 0];
    d1 += xv * lin_w[t * GG + 1];
    d2 += xv * lin_w[t * GG + 2];
  }
  for (int m = 16; m; m >>= 1) {
    d0 += __shfl_xor(d0, m, 32);
    d1 += __shfl_xor(d1, m, 32);
    d2 += __shfl_xor(d2, m, 32);
  }
  if (lane == 0) {
    d0 += lin_b[0]; d1 += lin_b[1]; d2 += lin_b[2];
    float mx = fmaxf(d0, fmaxf(d1, d2));
    float e0 = __expf(d0 - mx), e1 = __expf(d1 - mx), e2 = __expf(d2 - mx);
    float inv = 1.0f / (e0 + e1 + e2);
    s[node * GG + 0] = e0 * inv;
    s[node * GG + 1] = e1 * inv;
    s[node * GG + 2] = e2 * inv;
  }
}

__global__ __launch_bounds__(256)
void dgn_stats_kernel(const float* __restrict__ h, const float* __restrict__ s,
                      float* __restrict__ meanA, float* __restrict__ rstdA) {
  __shared__ float r1[256], r2[256];
  int f   = blockIdx.x;        // 0 .. G*HD-1
  int g   = f >> 7;
  int col = f & 127;
  int tid = threadIdx.x;
  float sum = 0.f, sq = 0.f;
  for (int n = tid; n < NN; n += 256) {
    float v = s[n * GG + g] * h[(size_t)n * HD + col];
    sum += v; sq += v * v;
  }
  r1[tid] = sum; r2[tid] = sq;
  __syncthreads();
  for (int st = 128; st; st >>= 1) {
    if (tid < st) { r1[tid] += r1[tid + st]; r2[tid] += r2[tid + st]; }
    __syncthreads();
  }
  if (tid == 0) {
    float mean = r1[0] / (float)NN;
    float var  = r2[0] / (float)NN - mean * mean;
    meanA[f] = mean;
    rstdA[f] = rsqrtf(var + EPSV);
  }
}

__global__ void dgn_apply_kernel(const float* __restrict__ h,
                                 const float* __restrict__ s,
                                 const float* __restrict__ meanA,
                                 const float* __restrict__ rstdA,
                                 const float* __restrict__ bn_w,
                                 const float* __restrict__ bn_b,
                                 uint16_t* __restrict__ h2b) {
  int idx = blockIdx.x * blockDim.x + threadIdx.x;   // NN * HD
  if (idx >= NN * HD) return;
  int n   = idx >> 7;
  int col = idx & 127;
  float xv = h[idx];
  float a = 0.f;
#pragma unroll
  for (int g = 0; g < GG; ++g) {
    int f = g * HD + col;
    float v = s[n * GG + g] * xv;
    a += (v - meanA[f]) * rstdA[f] * bn_w[f] + bn_b[f];
  }
  float r = fmaxf(xv + LAMDA * a, 0.0f);
  h2b[idx] = f2bf(r);
}

// ---------------------------------------------------------------------------
// Workspace layout (bytes, all 256B-aligned)
// ---------------------------------------------------------------------------
constexpr size_t XB_OFF   = 0;                               // NN*FIN bf16
constexpr size_t BS_OFF   = XB_OFF  + (size_t)NN * FIN * 2;  // NS*FIN bf16 (transposed stack)
constexpr size_t Y_OFF    = BS_OFF  + (size_t)FIN * NS * 2;  // NN*NS f32
constexpr size_t CNT_OFF  = Y_OFF   + (size_t)NN * NS * 4;   // NN*RR f32
constexpr size_t ACC_OFF  = CNT_OFF + (size_t)NN * RR * 4;   // NN*HD f32
constexpr size_t H1_OFF   = ACC_OFF + (size_t)NN * HD * 4;   // NN*HD f32
constexpr size_t H1B_OFF  = H1_OFF  + (size_t)NN * HD * 4;   // NN*HD bf16
constexpr size_t S_OFF    = H1B_OFF + (size_t)NN * HD * 2;   // NN*GG f32
constexpr size_t MEAN_OFF = S_OFF   + (size_t)NN * GG * 4;   // GG*HD f32
constexpr size_t RSTD_OFF = MEAN_OFF + 1536;                 // GG*HD f32

extern "C" void kernel_launch(void* const* d_in, const int* in_sizes, int n_in,
                              void* d_out, int out_size, void* d_ws, size_t ws_size,
                              hipStream_t stream) {
  const float* x          = (const float*)d_in[0];
  const int*   edge_index = (const int*)  d_in[1];
  const int*   edge_type  = (const int*)  d_in[2];
  const float* W1         = (const float*)d_in[3];
  const float* root1      = (const float*)d_in[4];
  const float* b1         = (const float*)d_in[5];
  const float* lin_w      = (const float*)d_in[6];
  const float* lin_b      = (const float*)d_in[7];
  const float* bn_w       = (const float*)d_in[8];
  const float* bn_b       = (const float*)d_in[9];
  const float* W2         = (const float*)d_in[10];
  const float* root2      = (const float*)d_in[11];
  const float* b2         = (const float*)d_in[12];
  float* out = (float*)d_out;

  char* ws = (char*)d_ws;
  uint16_t* Xb    = (uint16_t*)(ws + XB_OFF);
  uint16_t* Bt    = (uint16_t*)(ws + BS_OFF);
  float*    Y     = (float*)   (ws + Y_OFF);
  float*    cnt   = (float*)   (ws + CNT_OFF);
  float*    acc   = (float*)   (ws + ACC_OFF);
  float*    h1    = (float*)   (ws + H1_OFF);
  uint16_t* h1b   = (uint16_t*)(ws + H1B_OFF);
  float*    sG    = (float*)   (ws + S_OFF);
  float*    meanA = (float*)   (ws + MEAN_OFF);
  float*    rstdA = (float*)   (ws + RSTD_OFF);

  const dim3 blk(256);
  const dim3 gblk(128);                 // GEMM: 4 waves per block
  const dim3 ggrid(NS / BN, NN / BM);   // 9 x 32

  // ---- edge segment counts (shared by both layers) ----
  zero_f32_kernel<<<(NN * RR + 255) / 256, blk, 0, stream>>>(cnt, NN * RR);
  count_edges_kernel<<<EE / 256, blk, 0, stream>>>(edge_index, edge_type, cnt);

  // ---- layer 1: Y = Xbf16 @ [W1_0..W1_7 | root1]  (4096x4096 @ 4096x1152) ----
  cast_bf16_kernel<<<(NN * FIN / 4 + 255) / 256, blk, 0, stream>>>(x, Xb, NN * FIN / 4);
  pack_stackT_kernel<<<((size_t)FIN * NS + 255) / 256, blk, 0, stream>>>(W1, root1, Bt, FIN);
  gemm_bf16_wmma_kernel<<<ggrid, gblk, 0, stream>>>(Xb, Bt, Y, FIN);

  // scatter-mean in embedded space + root + bias
  zero_f32_kernel<<<(NN * HD + 255) / 256, blk, 0, stream>>>(acc, NN * HD);
  scatter_kernel<<<(EE * 32) / 256, blk, 0, stream>>>(Y, edge_index, edge_type, cnt, acc);
  finish_kernel<<<(NN * HD + 255) / 256, blk, 0, stream>>>(acc, Y, b1, h1);

  // ---- DiffGroupNorm + ReLU (produces bf16 activations for layer 2) ----
  dgn_softmax_kernel<<<NN / 8, blk, 0, stream>>>(h1, lin_w, lin_b, sG);
  dgn_stats_kernel<<<GG * HD, blk, 0, stream>>>(h1, sG, meanA, rstdA);
  dgn_apply_kernel<<<(NN * HD + 255) / 256, blk, 0, stream>>>(h1, sG, meanA, rstdA, bn_w, bn_b, h1b);

  // ---- layer 2: Y = h1b @ [W2_0..W2_7 | root2]  (4096x128 @ 128x1152) ----
  pack_stackT_kernel<<<((size_t)HD * NS + 255) / 256, blk, 0, stream>>>(W2, root2, Bt, HD);
  gemm_bf16_wmma_kernel<<<ggrid, gblk, 0, stream>>>(h1b, Bt, Y, HD);

  zero_f32_kernel<<<(NN * HD + 255) / 256, blk, 0, stream>>>(acc, NN * HD);
  scatter_kernel<<<(EE * 32) / 256, blk, 0, stream>>>(Y, edge_index, edge_type, cnt, acc);
  finish_kernel<<<(NN * HD + 255) / 256, blk, 0, stream>>>(acc, Y, b2, out);
}
